// Constrained_LCBlock_60988535603783
// MI455X (gfx1250) — matrix operand order repaired and checked by reading, not compiled, not run on hardware
//
#include <hip/hip_runtime.h>

typedef __attribute__((ext_vector_type(16))) _Float16 v16h;
typedef __attribute__((ext_vector_type(8)))  float    v8f;
typedef __attribute__((ext_vector_type(4)))  float    v4f;

namespace {
constexpr int CIN  = 32;
constexpr int COUT = 32;
constexpr int HH   = 64;
constexpr int WW   = 64;
constexpr int KPIX = CIN * 9;   // 288 reduction length per pixel
}

// One wave (32 lanes) owns one output pixel (h,w):
//   D[32(cout) x 16(batch)] = Wblk[32 x 288] * Patch[288 x 16], then ReLU.
// f16 WMMA 16x16x32, fp32 accumulation, kk loop fully unrolled so all
// k -> (cin,kh,kw) decompositions are compile-time (selected by lane-half).
__global__ __launch_bounds__(256) void lc_wmma_kernel(
    const float* __restrict__ x,     // [16][32][64][64]
    const float* __restrict__ wts,   // [64][64][32][32][3][3]
    float* __restrict__ out) {       // [16][32][64][64]
  const int lane = threadIdx.x & 31;
  const int wave = threadIdx.x >> 5;
  const int pix  = blockIdx.x * 8 + wave;   // 512 blocks * 8 waves = 4096 px
  const int ph   = pix >> 6;
  const int pw   = pix & 63;
  const int hl   = lane & 15;               // M-row / N-col / batch id
  const bool g   = (lane & 16) != 0;        // lane half (K sub-range select)

  // ---- wave-uniform 3x3 neighborhood: clamped offsets + validity ----
  int  rowoff[3], coloff[3];
  bool rok[3], cok[3];
#pragma unroll
  for (int i = 0; i < 3; ++i) {
    const int r = ph + i - 1;
    rok[i]    = (unsigned)r < (unsigned)HH;
    rowoff[i] = (r < 0 ? 0 : (r > HH - 1 ? HH - 1 : r)) * WW;
    const int c = pw + i - 1;
    cok[i]    = (unsigned)c < (unsigned)WW;
    coloff[i] = (c < 0 ? 0 : (c > WW - 1 ? WW - 1 : c));
  }
  int  rc[9];   // clamped (row*W + col) for the 9 taps (compile-time indexed)
  bool ok9[9];  // tap validity (zero-padding mask)
#pragma unroll
  for (int i = 0; i < 3; ++i)
#pragma unroll
    for (int j = 0; j < 3; ++j) {
      rc[i * 3 + j]  = rowoff[i] + coloff[j];
      ok9[i * 3 + j] = rok[i] && cok[j];
    }

  const int xlane = hl * (CIN * HH * WW);   // batch base (elements)
  const float* __restrict__ wblk = wts + (size_t)pix * COUT * KPIX;

  v8f acc[2] = {};  // two 16x16 f32 accumulator tiles (cout 0..15 / 16..31)

#pragma unroll
  for (int kk = 0; kk < 9; ++kk) {
    // ---- B fragment: Patch[K=288][N=16], N = batch.
    // lanes 0-15 hold K = kk*32+0..15, lanes 16-31 hold K = kk*32+16..31.
    v16h bfrag;
#pragma unroll
    for (int t = 0; t < 16; ++t) {
      const int ka = kk * 32 + t;       // K for lane-half 0 (compile-time)
      const int kb = ka + 16;           // K for lane-half 1 (compile-time)
      const int c0 = ka / 9, r0 = ka % 9;   // k = cin*9 + (kh*3+kw)
      const int c1 = kb / 9, r1 = kb % 9;
      const int a0 = rc[r0] + c0 * (HH * WW);
      const int a1 = rc[r1] + c1 * (HH * WW);
      const int addr = xlane + (g ? a1 : a0);     // always in-bounds (clamped)
      float v = x[addr];                          // unconditional, batched
      v = (g ? ok9[r1] : ok9[r0]) ? v : 0.0f;     // zero 'same' padding
      bfrag[t] = (_Float16)v;
    }

    // ---- A fragments + WMMA for both cout tiles.
    // A order used here (not AMD's table order): one run of 16 consecutive K per lane, at
    // (grp*16) within the 32-wide K step (matches B).
#pragma unroll
    for (int t = 0; t < 2; ++t) {
      const float* arow =
          wblk + (size_t)(t * 16 + hl) * KPIX + kk * 32 + (g ? 16 : 0);
      // weights are single-use: stream nontemporally, keep x in L2
      v4f a0 = __builtin_nontemporal_load((const v4f*)(arow + 0));
      v4f a1 = __builtin_nontemporal_load((const v4f*)(arow + 4));
      v4f a2 = __builtin_nontemporal_load((const v4f*)(arow + 8));
      v4f a3 = __builtin_nontemporal_load((const v4f*)(arow + 12));
      v16h afrag;
#pragma unroll
      for (int i = 0; i < 4; ++i) {
        afrag[i]      = (_Float16)a0[i];
        afrag[4 + i]  = (_Float16)a1[i];
        afrag[8 + i]  = (_Float16)a2[i];
        afrag[12 + i] = (_Float16)a3[i];
      }
      // (neg_a, A, neg_b, B, c_mod, C, reuse_a, reuse_b)
      acc[t] = __builtin_amdgcn_wmma_f32_16x16x32_f16(
          false, afrag, false, bfrag, (short)0, acc[t], false, false);
    }
  }

  // ---- ReLU + store. C/D layout: VGPR r -> M = r + 8*(lane>=16), N = hl.
#pragma unroll
  for (int t = 0; t < 2; ++t) {
#pragma unroll
    for (int r = 0; r < 8; ++r) {
      const int co  = t * 16 + (g ? 8 : 0) + r;
      const float v = fmaxf(acc[t][r], 0.0f);
      float* dst = &out[(((size_t)hl * COUT + co) * HH + ph) * WW + pw];
      __builtin_nontemporal_store(v, dst);
    }
  }
}

extern "C" void kernel_launch(void* const* d_in, const int* in_sizes, int n_in,
                              void* d_out, int out_size, void* d_ws, size_t ws_size,
                              hipStream_t stream) {
  const float* x   = (const float*)d_in[0];   // [16,32,64,64] fp32
  const float* wts = (const float*)d_in[1];   // [64,64,32,32,3,3] fp32
  float* out = (float*)d_out;                 // [16,32,64,64] fp32
  (void)in_sizes; (void)n_in; (void)out_size; (void)d_ws; (void)ws_size;
  dim3 grid(512), block(256);
  hipLaunchKernelGGL(lc_wmma_kernel, grid, block, 0, stream, x, wts, out);
}
